// KernelKMEAttention_90005334655139
// MI455X (gfx1250) — compile-verified, matrix operand order checked by reading, not run on hardware
//
#include <hip/hip_runtime.h>
#include <hip/hip_bf16.h>

// ---------------------------------------------------------------------------
// KME attention for MI455X (gfx1250): wave32 + v_wmma_f32_16x16x32_f16
// + Tensor Data Mover (TDM) double-buffered K/V staging into LDS.
// ---------------------------------------------------------------------------
typedef __attribute__((ext_vector_type(16))) _Float16     v16h;
typedef __attribute__((ext_vector_type(8)))  _Float16     v8h;
typedef __attribute__((ext_vector_type(8)))  float        v8f;
typedef __attribute__((ext_vector_type(4)))  unsigned int u32x4;
typedef __attribute__((ext_vector_type(8)))  unsigned int u32x8;

#define WMMA_F16(A, B, C) \
  __builtin_amdgcn_wmma_f32_16x16x32_f16(false, (A), false, (B), (short)0, (C), false, false)

// Sizes (compile-time constants of the problem)
#define BB  2
#define SS  1024
#define MM  8
#define DD  64
#define HH  8
#define NF  64
#define F2  128          // 2*NF
#define HD  512          // H*D
#define RR  16384        // B*S*M

// f16 operand chunk packers -------------------------------------------------
// A/B 16-bit operand layout for 16x16x32: lane ln (0..15) + half hi:
//   elements 0..7  <- K = hi*8 + j        (+ 32*chunk)
//   elements 8..15 <- K = 16 + hi*8 + j   (+ 32*chunk)
__device__ __forceinline__ v16h pack_f32_pair(const float* p0, const float* p1) {
  float4 x0 = *(const float4*)(p0);
  float4 x1 = *(const float4*)(p0 + 4);
  float4 x2 = *(const float4*)(p1);
  float4 x3 = *(const float4*)(p1 + 4);
  v16h r;
  r[0]=(_Float16)x0.x;  r[1]=(_Float16)x0.y;  r[2]=(_Float16)x0.z;  r[3]=(_Float16)x0.w;
  r[4]=(_Float16)x1.x;  r[5]=(_Float16)x1.y;  r[6]=(_Float16)x1.z;  r[7]=(_Float16)x1.w;
  r[8]=(_Float16)x2.x;  r[9]=(_Float16)x2.y;  r[10]=(_Float16)x2.z; r[11]=(_Float16)x2.w;
  r[12]=(_Float16)x3.x; r[13]=(_Float16)x3.y; r[14]=(_Float16)x3.z; r[15]=(_Float16)x3.w;
  return r;
}

__device__ __forceinline__ v16h pack_f32_chunk(const float* base, int kb0, int kb1) {
  return pack_f32_pair(base + kb0, base + kb1);
}

__device__ __forceinline__ v16h pack_f16_pair(const _Float16* p0, const _Float16* p1) {
  v8h a = *(const v8h*)p0;
  v8h b = *(const v8h*)p1;
  v16h r;
#pragma unroll
  for (int j = 0; j < 8; ++j) { r[j] = a[j]; r[8 + j] = b[j]; }
  return r;
}

__device__ __forceinline__ v16h pack_f32_strided(const float* base, int kb0, int kb1,
                                                 int stride, float scale) {
  v16h r;
#pragma unroll
  for (int j = 0; j < 8; ++j) {
    r[j]     = (_Float16)(base[(kb0 + j) * stride] * scale);
    r[8 + j] = (_Float16)(base[(kb1 + j) * stride] * scale);
  }
  return r;
}

// ---------------------------------------------------------------------------
// TDM: issue a 2D tensor_load_to_lds (f16 elements). Descriptor per
// CDNA5 ISA: group0 = {count, lds_addr, global_addr,
// type=2}; group1 = {data_size=2B, tensor_dim0/1, tile_dim0/1, dim0 stride}.
// Wave-uniform inputs; EXEC is ignored by TDM. Tracked with TENSORcnt.
// ---------------------------------------------------------------------------
__device__ __forceinline__ void tdm_load_2d_f16(unsigned lds_off,
                                                unsigned long long gaddr,
                                                unsigned tensor_d0, unsigned tensor_d1,
                                                unsigned tile_d0, unsigned tile_d1,
                                                unsigned long long stride0) {
  u32x4 g0;
  g0[0] = 1u;                                              // count=1 (valid user D#)
  g0[1] = lds_off;                                         // lds_addr [63:32]
  g0[2] = (unsigned)(gaddr & 0xFFFFFFFFu);                 // global_addr [95:64]
  g0[3] = (unsigned)((gaddr >> 32) & 0x1FFFFFFu)           // global_addr [120:96]
        | (2u << 30);                                      // type=2 ("image")
  u32x8 g1;
  g1[0] = (1u << 16);                                      // data_size=1 (2B), wg_mask=0
  g1[1] = (tensor_d0 & 0xFFFFu) << 16;                     // tensor_dim0 lo16 @bit48
  g1[2] = (tensor_d0 >> 16) | ((tensor_d1 & 0xFFFFu) << 16);
  g1[3] = (tensor_d1 >> 16) | (tile_d0 << 16);             // tile_dim0 @bit112
  g1[4] = tile_d1;                                         // tile_dim1, tile_dim2=0
  g1[5] = (unsigned)(stride0 & 0xFFFFFFFFu);               // tensor_dim0_stride lo32
  g1[6] = (unsigned)((stride0 >> 32) & 0xFFFFu);           // stride hi16, stride1=0
  g1[7] = 0u;
  asm volatile("tensor_load_to_lds %0, %1" :: "s"(g0), "s"(g1) : "memory");
}

// ---------------------------------------------------------------------------
// Kernel 1: Q/K/V projection GEMM: Y(16384x512) = X(16384x64) @ W^T(64x512).
// Q,K stored f32 row-major; V stored transposed f16: vaT[(b*H+h)*512+col][s].
// ---------------------------------------------------------------------------
__global__ __launch_bounds__(128) void kme_proj(
    const float* __restrict__ qA, const float* __restrict__ kA, const float* __restrict__ vA,
    const float* __restrict__ Wq, const float* __restrict__ Wk, const float* __restrict__ Wv,
    float* __restrict__ pq, float* __restrict__ pk, _Float16* __restrict__ vaT) {
  const int which = blockIdx.y;
  const float* X = (which == 0) ? qA : ((which == 1) ? kA : vA);
  const float* W = (which == 0) ? Wq : ((which == 1) ? Wk : Wv);

  const int lane = threadIdx.x & 31;
  const int wv   = threadIdx.x >> 5;
  const int rt   = blockIdx.x * 4 + wv;      // row tile 0..1023
  const int ln   = lane & 15;
  const int hi   = lane >> 4;
  const int row0 = rt * 16;
  const int kb0  = hi * 8;

  // A operand: this wave's 16x64 row block, two K32 chunks
  const float* xr = X + (size_t)(row0 + ln) * DD;
  v16h a0 = pack_f32_chunk(xr, kb0, kb0 + 16);
  v16h a1 = pack_f32_chunk(xr, kb0 + 32, kb0 + 48);

  for (int nt = 0; nt < 32; ++nt) {
    const int col = nt * 16 + ln;
    const float* wr = W + (size_t)col * DD;
    v16h b0 = pack_f32_chunk(wr, kb0, kb0 + 16);
    v16h b1 = pack_f32_chunk(wr, kb0 + 32, kb0 + 48);
    v8f acc = {};
    acc = WMMA_F16(a0, b0, acc);
    acc = WMMA_F16(a1, b1, acc);

    if (which < 2) {
      float* out = (which == 0) ? pq : pk;
#pragma unroll
      for (int r = 0; r < 8; ++r) {
        const int rowg = row0 + r + hi * 8;      // D-tile: VGPR r -> row r (+8 for hi half)
        out[(size_t)rowg * HD + col] = acc[r];
      }
    } else {
      const int h = col >> 6, d = col & 63;
#pragma unroll
      for (int r = 0; r < 8; ++r) {
        const int rowg = row0 + r + hi * 8;
        const int b = rowg >> 13, s = (rowg >> 3) & (SS - 1), m = rowg & 7;
        vaT[((size_t)(b * HH + h) * HD + m * DD + d) * SS + s] = (_Float16)acc[r];
      }
    }
  }
}

// ---------------------------------------------------------------------------
// Kernel 2: RoPE + RFF encode. One wave per (row-tile of 16 = 2 s-values, head,
// q-or-k). RoPE'd rows staged to LDS as f16, WMMA vs per-head scaled freq,
// cos/sin epilogue with the atom-softmax sum folded into the 8 accumulator
// VGPRs (VGPR r <-> atom m=r for this lane's s).
// ---------------------------------------------------------------------------
__global__ __launch_bounds__(32) void kme_rff(
    const float* __restrict__ pq, const float* __restrict__ pk,
    const float* __restrict__ qlw, const float* __restrict__ klw,
    const float* __restrict__ cosT, const float* __restrict__ sinT,
    const float* __restrict__ rff, const float* __restrict__ logbw,
    _Float16* __restrict__ qr, _Float16* __restrict__ kr) {
  __shared__ __align__(16) _Float16 arope[16 * 64];

  const int rt = blockIdx.x, h = blockIdx.y, qk = blockIdx.z;
  const float*    P   = (qk == 0) ? pq : pk;
  const float*    LW  = (qk == 0) ? qlw : klw;
  _Float16*       OUT = (qk == 0) ? qr : kr;

  const int lane = threadIdx.x & 31;
  const int ln = lane & 15, hi = lane >> 4;
  const int row0 = rt * 16;

  // Stage RoPE'd 16x64 tile into LDS (lane -> row ln, half hi -> 32 columns)
  {
    const int rowg = row0 + ln;
    const int s = (rowg >> 3) & (SS - 1);
    const float* ar = P + (size_t)rowg * HD + h * DD;
    const float* cr = cosT + (size_t)s * DD;
    const float* sr = sinT + (size_t)s * DD;
#pragma unroll 8
    for (int c = hi * 32; c < hi * 32 + 32; ++c) {
      const float v   = ar[c];
      const float rot = (c < 32) ? -ar[c + 32] : ar[c - 32];
      arope[ln * 64 + c] = (_Float16)(v * cr[c] + rot * sr[c]);
    }
  }
  __syncthreads();

  // Atom softmax weights for this half-lane's s
  const int bb  = row0 >> 13;
  const int s_l = ((rt * 2) & (SS - 1)) + hi;
  float w[8];
  {
    const float* lwp = LW + ((size_t)bb * SS + s_l) * MM;
    float mx = -1e30f;
#pragma unroll
    for (int m = 0; m < 8; ++m) { w[m] = lwp[m]; mx = fmaxf(mx, w[m]); }
    float sum = 0.f;
#pragma unroll
    for (int m = 0; m < 8; ++m) { w[m] = expf(w[m] - mx); sum += w[m]; }
    const float inv = 1.0f / sum;
#pragma unroll
    for (int m = 0; m < 8; ++m) w[m] *= inv;
  }

  const float invbw = expf(-logbw[h]);
  const float* Fh = rff + (size_t)h * DD * NF;

  const int kb0 = hi * 8;
  const _Float16* arow = arope + ln * 64;
  v16h a0 = pack_f16_pair(arow + kb0,      arow + kb0 + 16);
  v16h a1 = pack_f16_pair(arow + kb0 + 32, arow + kb0 + 48);

  for (int nf = 0; nf < 4; ++nf) {
    const int f = nf * 16 + ln;
    v16h b0 = pack_f32_strided(Fh + f, kb0,      kb0 + 16, NF, invbw);
    v16h b1 = pack_f32_strided(Fh + f, kb0 + 32, kb0 + 48, NF, invbw);
    v8f pr = {};
    pr = WMMA_F16(a0, b0, pr);
    pr = WMMA_F16(a1, b1, pr);

    float cs = 0.f, sn = 0.f;
#pragma unroll
    for (int m = 0; m < 8; ++m) {          // VGPR m == atom m: softmax-weighted sum
      const float p = pr[m];
      cs += cosf(p) * w[m];
      sn += sinf(p) * w[m];
    }
    const float rscale = 0.125f;           // 1/sqrt(NF)
    const size_t off = (((size_t)bb * HH + h) * SS + s_l) * F2;
    OUT[off + f]      = (_Float16)(cs * rscale);
    OUT[off + NF + f] = (_Float16)(sn * rscale);
  }
}

// ---------------------------------------------------------------------------
// Kernel 3: flash attention with TDM-staged K/V tiles.
// Block = 4 waves; wave = (16 q rows) x (128 of 512 PV columns).
// Per 32-k step: TDM double-buffers the 32x128 f16 k_rff tile (8 KB) and the
// 512x32 f16 vaT tile (32 KB) into LDS (wave 0 issues, TENSORcnt + barrier
// hand-off); 8 score WMMAs, online softmax with shfl_xor row reductions,
// P transposed via LDS, 8 PV WMMAs. 64 f32 accumulator VGPRs per wave.
// LDS: 2*(8+32) KB + 4 KB = 84 KB of the 320 KB WGP pool.
// ---------------------------------------------------------------------------
__global__ __launch_bounds__(128) void kme_attn(
    const _Float16* __restrict__ qr, const _Float16* __restrict__ kr,
    const _Float16* __restrict__ vaT, float* __restrict__ attn_out) {
  __shared__ __align__(16) _Float16 ksh[2][32 * F2];   // 16 KB (double-buffered)
  __shared__ __align__(16) _Float16 vsh[2][HD * 32];   // 64 KB (double-buffered)
  __shared__ __align__(16) _Float16 psh[4][16 * 32];   //  4 KB (P transpose)

  const int qt = blockIdx.x;                 // q tile (0..63)
  const int h = blockIdx.y, b = blockIdx.z;
  const int lane = threadIdx.x & 31;
  const int wv = threadIdx.x >> 5;
  const int ln = lane & 15, hi = lane >> 4;
  const int colw = wv * 128;
  const int kb0 = hi * 8;

  const size_t bh = (size_t)b * HH + h;
  const _Float16* qbase = qr + bh * SS * F2;
  const unsigned long long kg = (unsigned long long)(uintptr_t)(kr + bh * SS * F2);
  const unsigned long long vg = (unsigned long long)(uintptr_t)(vaT + bh * HD * SS);

  // A operand: q_rff rows, 4 K32 chunks covering F=128 (loaded once)
  v16h qa[4];
  {
    const _Float16* qrow = qbase + (size_t)(qt * 16 + ln) * F2;
#pragma unroll
    for (int c = 0; c < 4; ++c)
      qa[c] = pack_f16_pair(qrow + c * 32 + kb0, qrow + c * 32 + kb0 + 16);
  }

  v8f oacc[8];
#pragma unroll
  for (int i = 0; i < 8; ++i) oacc[i] = (v8f){};
  float mrun[8], lrun[8];
#pragma unroll
  for (int r = 0; r < 8; ++r) { mrun[r] = -1e30f; lrun[r] = 0.f; }

  // Prologue: stage k-chunk 0 into buffer 0.
  if (wv == 0) {
    // k_rff tile: dim0 = f (contig, 128), dim1 = 32 k-rows, row stride F2
    tdm_load_2d_f16((unsigned)(uintptr_t)&ksh[0][0], kg,
                    F2, SS, F2, 32, F2);
    // vaT tile: dim0 = k (contig, 32 of S), dim1 = 512 cols, col stride SS
    tdm_load_2d_f16((unsigned)(uintptr_t)&vsh[0][0], vg,
                    SS, HD, 32, HD, SS);
  }

  int buf = 0;
  for (int kc = 0; kc < SS / 32; ++kc) {
    // Issue next chunk into the other buffer, then wait for current chunk.
    if (wv == 0) {
      if (kc + 1 < SS / 32) {
        const unsigned long long koff = (unsigned long long)(kc + 1) * 32ull;
        tdm_load_2d_f16((unsigned)(uintptr_t)&ksh[buf ^ 1][0], kg + koff * F2 * 2ull,
                        F2, SS, F2, 32, F2);
        tdm_load_2d_f16((unsigned)(uintptr_t)&vsh[buf ^ 1][0], vg + koff * 2ull,
                        SS, HD, 32, HD, SS);
        __builtin_amdgcn_s_wait_tensorcnt((short)2);   // in-order: current pair done
      } else {
        __builtin_amdgcn_s_wait_tensorcnt((short)0);
      }
    }
    __syncthreads();                                   // release consumers

    // --- scores: two adjacent 16x16 tiles from the staged k-tile
    v8f sc0 = {}, sc1 = {};
    {
      const _Float16* kr0 = &ksh[buf][(size_t)ln * F2];
      const _Float16* kr1 = &ksh[buf][(size_t)(16 + ln) * F2];
#pragma unroll
      for (int c = 0; c < 4; ++c) {
        v16h bt = pack_f16_pair(kr0 + c * 32 + kb0, kr0 + c * 32 + kb0 + 16);
        sc0 = WMMA_F16(qa[c], bt, sc0);
      }
#pragma unroll
      for (int c = 0; c < 4; ++c) {
        v16h bt = pack_f16_pair(kr1 + c * 32 + kb0, kr1 + c * 32 + kb0 + 16);
        sc1 = WMMA_F16(qa[c], bt, sc1);
      }
    }

    // --- online softmax (per accumulator row; rows replicated across 16 lanes)
    _Float16* pp = psh[wv];
#pragma unroll
    for (int r = 0; r < 8; ++r) {
      float t = fmaxf(sc0[r], sc1[r]);
#pragma unroll
      for (int m = 1; m < 16; m <<= 1) t = fmaxf(t, __shfl_xor(t, m, 32));
      const float mnew = fmaxf(mrun[r], t);
      const float p0 = expf(sc0[r] - mnew);
      const float p1 = expf(sc1[r] - mnew);
      float rs = p0 + p1;
#pragma unroll
      for (int m = 1; m < 16; m <<= 1) rs += __shfl_xor(rs, m, 32);
      const float corr = expf(mrun[r] - mnew);
      lrun[r] = lrun[r] * corr + rs;
      mrun[r] = mnew;
#pragma unroll
      for (int nt = 0; nt < 8; ++nt) oacc[nt][r] *= corr;
      const int row = r + hi * 8;
      pp[row * 32 + ln]      = (_Float16)p0;
      pp[row * 32 + 16 + ln] = (_Float16)p1;
    }
    __syncthreads();

    // --- transpose P through LDS into the A operand layout (16x32, K=32)
    v16h pA = pack_f16_pair(pp + ln * 32 + kb0, pp + ln * 32 + kb0 + 16);

    // --- PV: 8 column tiles of the staged transposed V (contiguous f16 in k)
#pragma unroll
    for (int nt = 0; nt < 8; ++nt) {
      const int col = colw + nt * 16 + ln;
      const _Float16* vcol = &vsh[buf][(size_t)col * 32];
      v16h vb = pack_f16_pair(vcol + kb0, vcol + kb0 + 16);
      oacc[nt] = WMMA_F16(pA, vb, oacc[nt]);
    }
    __syncthreads();                                   // buffer-reuse fence
    buf ^= 1;
  }

  // normalize and store out[(b,h,q), m*64+d]
  float* obase = attn_out + (bh * SS + (size_t)qt * 16) * HD;
#pragma unroll
  for (int r = 0; r < 8; ++r) {
    const float inv = 1.0f / lrun[r];
    const int row = r + hi * 8;
#pragma unroll
    for (int nt = 0; nt < 8; ++nt)
      obase[(size_t)row * HD + colw + nt * 16 + ln] = oacc[nt][r] * inv;
  }
}

// ---------------------------------------------------------------------------
// Kernel 4: output projection: out_atoms(16384x64) = out(16384x512) @ Wo^T.
// K=512 -> 16 K32 WMMA chunks. Writes straight to d_out.
// ---------------------------------------------------------------------------
__global__ __launch_bounds__(128) void kme_out(
    const float* __restrict__ attn_out, const float* __restrict__ Wo,
    float* __restrict__ out_atoms) {
  const int lane = threadIdx.x & 31;
  const int wv = threadIdx.x >> 5;
  const int rt = blockIdx.x * 4 + wv;        // 0..1023
  const int ln = lane & 15, hi = lane >> 4;
  const int row0 = rt * 16;

  const int rowg = row0 + ln;
  const int b = rowg >> 13, s = (rowg >> 3) & (SS - 1), m = rowg & 7;

  for (int nt = 0; nt < 4; ++nt) {
    const int d = nt * 16 + ln;
    v8f acc = {};
    for (int kc = 0; kc < 16; ++kc) {
      const int kg0 = kc * 32 + hi * 8;
      const int kg1 = kg0 + 16;
      // A element (rowg, k): k = h*64+dd -> attn[((b*8+h)*S+s)*512 + m*64+dd]
      const int h0 = kg0 >> 6, dd0 = kg0 & 63;
      const int h1 = kg1 >> 6, dd1 = kg1 & 63;
      const float* p0 = attn_out + (((size_t)b * HH + h0) * SS + s) * HD + m * DD + dd0;
      const float* p1 = attn_out + (((size_t)b * HH + h1) * SS + s) * HD + m * DD + dd1;
      v16h a = pack_f32_pair(p0, p1);
      v16h bt = pack_f32_chunk(Wo + (size_t)d * HD, kg0, kg1);
      acc = WMMA_F16(a, bt, acc);
    }
#pragma unroll
    for (int r = 0; r < 8; ++r) {
      const int rg = row0 + r + hi * 8;
      out_atoms[(size_t)rg * DD + d] = acc[r];
    }
  }
}

// ---------------------------------------------------------------------------
// Kernel 5: log-weight delta: lw = q_lw + Ww @ mean_m(out_atoms)
// ---------------------------------------------------------------------------
__global__ __launch_bounds__(256) void kme_delta(
    const float* __restrict__ out_atoms, const float* __restrict__ Ww,
    const float* __restrict__ qlw, float* __restrict__ lw_out) {
  const int t = blockIdx.x * blockDim.x + threadIdx.x;   // (b*S+s) 0..2047
  if (t >= BB * SS) return;
  const float* base = out_atoms + (size_t)t * MM * DD;
  float acc[8] = {0, 0, 0, 0, 0, 0, 0, 0};
  for (int d = 0; d < DD; ++d) {
    float mean = 0.f;
#pragma unroll
    for (int m = 0; m < 8; ++m) mean += base[m * DD + d];
    mean *= 0.125f;
#pragma unroll
    for (int mm = 0; mm < 8; ++mm) acc[mm] += mean * Ww[mm * DD + d];
  }
#pragma unroll
  for (int mm = 0; mm < 8; ++mm)
    lw_out[(size_t)t * MM + mm] = qlw[(size_t)t * MM + mm] + acc[mm];
}

// ---------------------------------------------------------------------------
extern "C" void kernel_launch(void* const* d_in, const int* in_sizes, int n_in,
                              void* d_out, int out_size, void* d_ws, size_t ws_size,
                              hipStream_t stream) {
  const float* q_atoms = (const float*)d_in[0];
  const float* qlw     = (const float*)d_in[1];
  const float* k_atoms = (const float*)d_in[2];
  const float* klw     = (const float*)d_in[3];
  const float* v_atoms = (const float*)d_in[4];
  // d_in[5] = v_log_weights (unused by the reference math)
  const float* cosT    = (const float*)d_in[6];
  const float* sinT    = (const float*)d_in[7];
  const float* Wq      = (const float*)d_in[8];
  const float* Wk      = (const float*)d_in[9];
  const float* Wv      = (const float*)d_in[10];
  const float* Wo      = (const float*)d_in[11];
  const float* Ww      = (const float*)d_in[12];
  const float* logbw   = (const float*)d_in[13];
  const float* rff     = (const float*)d_in[14];

  char* ws = (char*)d_ws;
  float*     pq   = (float*)(ws + 0);                    // 33,554,432 B
  float*     pk   = (float*)(ws + 33554432);             // 33,554,432 B
  _Float16*  vaT  = (_Float16*)(ws + 67108864);          // 16,777,216 B (f16, transposed)
  _Float16*  qrf  = (_Float16*)(ws + 83886080);          //  4,194,304 B
  _Float16*  krf  = (_Float16*)(ws + 88080384);          //  4,194,304 B
  float*     attn = (float*)(ws + 92274688);             // 33,554,432 B  (total ~120 MB)

  float* out_atoms = (float*)d_out;
  float* lw_out    = out_atoms + (size_t)BB * SS * MM * DD;

  kme_proj <<<dim3(256, 3, 1),  128, 0, stream>>>(q_atoms, k_atoms, v_atoms,
                                                  Wq, Wk, Wv, pq, pk, vaT);
  kme_rff  <<<dim3(1024, 8, 2),  32, 0, stream>>>(pq, pk, qlw, klw, cosT, sinT,
                                                  rff, logbw, qrf, krf);
  kme_attn <<<dim3(64, 8, 2),   128, 0, stream>>>(qrf, krf, vaT, attn);
  kme_out  <<<dim3(256, 1, 1),  128, 0, stream>>>(attn, Wo, out_atoms);
  kme_delta<<<dim3(8, 1, 1),    256, 0, stream>>>(out_atoms, Ww, qlw, lw_out);
}
